// MessageProp_76879914598905
// MI455X (gfx1250) — matrix operand order repaired and checked by reading, not compiled, not run on hardware
//
#include <hip/hip_runtime.h>
#include <hip/hip_bf16.h>

// ---------------------------------------------------------------------------
// MessageProp fused kernel for MI455X (gfx1250, wave32, WMMA bf16)
//   N=262144 rows, D=128, H=132 (padded to 160), K=8 components
//   Pipeline: sum_k comp -> @Wm(+8*bm) -> @Wu(+bu) -> concat(signal, .)
//             -> 4-layer FC stack (ReLU x3) -> out [N,128] f32
// HBM-bound: ~1.35 GB mandatory traffic @ 23.3 TB/s ~= 58 us; bf16 WMMA chain
// (~62 GFLOP) is far below the matrix-op ceiling, so everything is fused into
// one streaming pass with activations resident in LDS.
// ---------------------------------------------------------------------------

#define NROWS 262144
#define DD    128
#define HP    160   // H=132 padded to multiple of 32

typedef __attribute__((ext_vector_type(16))) __bf16 v16bf;
typedef __attribute__((ext_vector_type(8)))  float  v8f;
typedef __attribute__((ext_vector_type(4)))  unsigned int v4u;

// ws layout (bytes):
//   packed bf16 weights (per-lane contiguous 32B B-fragments):
//     Wm @ 0        (4x8   = 32 tiles, 32 KiB)
//     Wu @ 32768    (32 tiles)
//     W0 @ 65536    (8x10  = 80 tiles)
//     W1 @ 147456   (5x10  = 50 tiles)
//     W2 @ 198656   (50 tiles)
//     W3 @ 249856   (5x8   = 40 tiles)
//   biases f32 @ 290816: bmK[128] bu[128] b0[160] b1[160] b2[160] b3[128]
#define WOFF_WM 0
#define WOFF_WU 8192u
#define WOFF_W0 16384u
#define WOFF_W1 36864u
#define WOFF_W2 49664u
#define WOFF_W3 62464u
#define BIAS_BYTE_OFF 290816u

__device__ __forceinline__ unsigned short f32_to_bf16_u(float f) {
    unsigned int x = __float_as_uint(f);
    unsigned int r = x + 0x7FFFu + ((x >> 16) & 1u);   // round-to-nearest-even
    return (unsigned short)(r >> 16);
}
__device__ __forceinline__ unsigned int pack2_bf16(float a, float b) {
    return (unsigned int)f32_to_bf16_u(a) | ((unsigned int)f32_to_bf16_u(b) << 16);
}

// ---------------------------------------------------------------------------
// Prep: pack one weight matrix W[R][C] (row-major, x@W convention) into
// WMMA B-fragment tiles of 32(K) x 16(N), zero-padded to KT*32 x NT*16.
// Per tile: lane-major, 8 dwords per lane; dword j of lane l holds rows
// (8*(l>>4) + 2j + (j>=4 ? 8 : 0), +1) of column (l&15), packed bf16.
// ---------------------------------------------------------------------------
__global__ void pack_weights_kernel(const float* __restrict__ W, int R, int C,
                                    int NT, unsigned int* __restrict__ dst) {
    int t    = blockIdx.x * 256 + threadIdx.x;
    int tile = t >> 8;
    int lane = (t >> 3) & 31;
    int j    = t & 7;
    int kt = tile / NT;
    int nt = tile - kt * NT;
    int k = kt * 32 + ((lane >> 4) << 3) + 2 * j + ((j >> 2) << 3);
    int n = nt * 16 + (lane & 15);
    float lo = (k     < R && n < C) ? W[(size_t)k * C + n]       : 0.0f;
    float hi = (k + 1 < R && n < C) ? W[(size_t)(k + 1) * C + n] : 0.0f;
    dst[(size_t)tile * 256 + lane * 8 + j] = pack2_bf16(lo, hi);
}

__global__ void pack_bias_kernel(const float* bm, const float* bu,
                                 const float* b0, const float* b1,
                                 const float* b2, const float* b3,
                                 float* __restrict__ dst) {
    int t = blockIdx.x * blockDim.x + threadIdx.x;
    if (t < 128)      dst[t] = 8.0f * bm[t];                       // K*bm folded
    else if (t < 256) dst[t] = bu[t - 128];
    else if (t < 416) { int i = t - 256; dst[t] = (i < 132) ? b0[i] : 0.0f; }
    else if (t < 576) { int i = t - 416; dst[t] = (i < 132) ? b1[i] : 0.0f; }
    else if (t < 736) { int i = t - 576; dst[t] = (i < 132) ? b2[i] : 0.0f; }
    else if (t < 864) dst[t] = b3[t - 736];
}

// ---------------------------------------------------------------------------
// LDS layout (dynamic, 237568 B):
//   xbuf : bf16 [128][272]  (cols 0..255 used: [signal | right])   @ 0
//   h0   : bf16 [128][168]  (cols 0..159 used)                     @ 69632
//   h1   : bf16 [128][168]                                         @ 112640
//   wbuf : packed weight tiles, up to 80 tiles * 1 KiB             @ 155648
// ---------------------------------------------------------------------------
#define SM_X  0u
#define SM_H0 69632u
#define SM_H1 112640u
#define SM_W  155648u
#define SMEM_BYTES 237568u
#define XSTR 272
#define HSTR 168

// CDNA5 async global->LDS copy (ASYNCcnt path): 16 B per lane per issue, no
// VGPR staging. LDS address operand = low 32 bits of the generic pointer
__device__ __forceinline__ void stage_weights_async(
    const unsigned int* __restrict__ g, unsigned int* __restrict__ lds,
    int nUint) {
    int n4 = nUint >> 2;
    for (int i = threadIdx.x; i < n4; i += 256) {
        const unsigned int* gp = g + ((size_t)i << 2);
        unsigned int ldsAddr = (unsigned int)(size_t)(lds + ((size_t)i << 2));
        asm volatile("global_load_async_to_lds_b128 %0, %1, off"
                     :
                     : "v"(ldsAddr), "v"(gp)
                     : "memory");
    }
    asm volatile("s_wait_asynccnt 0x0" ::: "memory");
}

// One fused GEMM layer: act[128][Win=KT*32] (bf16, LDS) x W -> out (NT*16 wide)
template <int KT, int NT, bool RELU, bool TO_GLOBAL>
__device__ __forceinline__ void gemm_layer(
    const unsigned short* __restrict__ src, int srcStride,
    unsigned short* __restrict__ dst, int dstStride, int dstColOff,
    float* __restrict__ gout, int gRowBase,
    const unsigned int* __restrict__ wlds,
    const float* __restrict__ bias) {
    const int tid  = threadIdx.x;
    const int lane = tid & 31;
    const int m0   = (tid >> 5) << 4;          // 16-row stripe per wave

    v8f C[NT];
#pragma unroll
    for (int nt = 0; nt < NT; ++nt) C[nt] = v8f{};

    // A-fragment addressing (16-bit A 16x32 layout):
    //   lanes 0-15: M=lane,   K chunks [0..7] and [16..23]
    //   lanes16-31: M=lane-16,K chunks [8..15] and [24..31]
    const int mrow   = m0 + (lane & 15);
    const int kboff  = (lane >> 4) << 3;
    const unsigned short* arow = src + (size_t)mrow * srcStride;

    for (int kt = 0; kt < KT; ++kt) {
        union { v16bf v; v4u u[2]; } A;
        const unsigned short* ap = arow + kt * 32 + kboff;
        A.u[0] = *(const v4u*)(ap);
        A.u[1] = *(const v4u*)(ap + 16);
#pragma unroll
        for (int nt = 0; nt < NT; ++nt) {
            union { v16bf v; v4u u[2]; } B;
            const unsigned int* bp = wlds + (((size_t)(kt * NT + nt) * 32 + lane) << 3);
            B.u[0] = *(const v4u*)(bp);
            B.u[1] = *(const v4u*)(bp + 4);
            C[nt] = __builtin_amdgcn_wmma_f32_16x16x32_bf16(
                false, A.v, false, B.v, (short)0, C[nt], false, false);
        }
    }

    // Epilogue per the 32-bit C/D 16x16 layout:
    //   lanes 0-15: N=lane, VGPR r -> M=r ; lanes 16-31: N=lane-16, M=8+r
    const int ncol  = lane & 15;
    const int mhalf = (lane >> 4) << 3;
#pragma unroll
    for (int nt = 0; nt < NT; ++nt) {
        int n = nt * 16 + ncol;
        float b = bias[n];
#pragma unroll
        for (int r = 0; r < 8; ++r) {
            float v = C[nt][r] + b;
            if (RELU) v = fmaxf(v, 0.0f);
            int m = m0 + mhalf + r;
            if (TO_GLOBAL) {
                gout[(size_t)(gRowBase + m) * DD + n] = v;
            } else {
                dst[(size_t)m * dstStride + dstColOff + n] = f32_to_bf16_u(v);
            }
        }
    }
}

__global__ __launch_bounds__(256, 1) void messageprop_fused_kernel(
    const float* __restrict__ signal, const float* __restrict__ comps,
    const unsigned int* __restrict__ wpack, const float* __restrict__ bias,
    float* __restrict__ out) {
    extern __shared__ __align__(16) unsigned char smem[];
    unsigned short* xbuf = (unsigned short*)(smem + SM_X);
    unsigned short* h0   = (unsigned short*)(smem + SM_H0);
    unsigned short* h1   = (unsigned short*)(smem + SM_H1);
    unsigned int*   wbuf = (unsigned int*)(smem + SM_W);

    const int row0 = blockIdx.x * 128;
    const int tid  = threadIdx.x;

    // Kick off Wm+Wu staging (64 KiB combined fits the 80 KiB wbuf) ; the
    // async copy completes before the component streaming below.
    stage_weights_async(wpack + WOFF_WM, wbuf, 64 * 256);

    // ---- Phase 0: stream signal + sum of 8 components into LDS (bf16) ----
    for (int it = tid; it < 128 * 32; it += 256) {   // 32 float4 per row
        int r = it >> 5;
        int c = (it & 31) << 2;
        const float4 s = *(const float4*)(signal + (size_t)(row0 + r) * DD + c);
        uint2 ps;
        ps.x = pack2_bf16(s.x, s.y);
        ps.y = pack2_bf16(s.z, s.w);
        *(uint2*)(xbuf + (size_t)r * XSTR + c) = ps;

        float4 acc = {0.f, 0.f, 0.f, 0.f};
#pragma unroll
        for (int k = 0; k < 8; ++k) {
            const float4 v = *(const float4*)(comps +
                ((size_t)k * NROWS + row0 + r) * DD + c);
            acc.x += v.x; acc.y += v.y; acc.z += v.z; acc.w += v.w;
        }
        uint2 pa;
        pa.x = pack2_bf16(acc.x, acc.y);
        pa.y = pack2_bf16(acc.z, acc.w);
        *(uint2*)(h0 + (size_t)r * HSTR + c) = pa;
    }
    __syncthreads();

    // ---- right = summed @ Wm + 8*bm ----
    gemm_layer<4, 8, false, false>(h0, HSTR, h1, HSTR, 0, nullptr, 0,
                                   wbuf, bias + 0);
    __syncthreads();

    // ---- right = right @ Wu + bu  -> xbuf cols 128..255 ----
    gemm_layer<4, 8, false, false>(h1, HSTR, xbuf, XSTR, 128, nullptr, 0,
                                   wbuf + 32 * 256, bias + 128);
    __syncthreads();

    // ---- h = relu(x @ W0 + b0)  (256 -> 160) ----
    stage_weights_async(wpack + WOFF_W0, wbuf, 80 * 256);
    __syncthreads();
    gemm_layer<8, 10, true, false>(xbuf, XSTR, h0, HSTR, 0, nullptr, 0,
                                   wbuf, bias + 256);
    __syncthreads();

    // ---- h = relu(h @ W1 + b1) ----
    stage_weights_async(wpack + WOFF_W1, wbuf, 50 * 256);
    __syncthreads();
    gemm_layer<5, 10, true, false>(h0, HSTR, h1, HSTR, 0, nullptr, 0,
                                   wbuf, bias + 416);
    __syncthreads();

    // ---- h = relu(h @ W2 + b2) ----
    stage_weights_async(wpack + WOFF_W2, wbuf, 50 * 256);
    __syncthreads();
    gemm_layer<5, 10, true, false>(h1, HSTR, h0, HSTR, 0, nullptr, 0,
                                   wbuf, bias + 576);
    __syncthreads();

    // ---- out = h @ W3 + b3  (f32 to global) ----
    stage_weights_async(wpack + WOFF_W3, wbuf, 40 * 256);
    __syncthreads();
    gemm_layer<5, 8, false, true>(h0, HSTR, nullptr, 0, 0, out, row0,
                                  wbuf, bias + 736);
}

// ---------------------------------------------------------------------------
extern "C" void kernel_launch(void* const* d_in, const int* in_sizes, int n_in,
                              void* d_out, int out_size, void* d_ws, size_t ws_size,
                              hipStream_t stream) {
    const float* signal = (const float*)d_in[0];
    const float* comps  = (const float*)d_in[1];
    const float* Wm = (const float*)d_in[2];
    const float* bm = (const float*)d_in[3];
    const float* Wu = (const float*)d_in[4];
    const float* bu = (const float*)d_in[5];
    const float* W0 = (const float*)d_in[6];
    const float* b0 = (const float*)d_in[7];
    const float* W1 = (const float*)d_in[8];
    const float* b1 = (const float*)d_in[9];
    const float* W2 = (const float*)d_in[10];
    const float* b2 = (const float*)d_in[11];
    const float* W3 = (const float*)d_in[12];
    const float* b3 = (const float*)d_in[13];

    unsigned int* wpack = (unsigned int*)d_ws;
    float* biasws = (float*)((unsigned char*)d_ws + BIAS_BYTE_OFF);

    pack_weights_kernel<<<32, 256, 0, stream>>>(Wm, 128, 128,  8, wpack + WOFF_WM);
    pack_weights_kernel<<<32, 256, 0, stream>>>(Wu, 128, 128,  8, wpack + WOFF_WU);
    pack_weights_kernel<<<80, 256, 0, stream>>>(W0, 256, 132, 10, wpack + WOFF_W0);
    pack_weights_kernel<<<50, 256, 0, stream>>>(W1, 132, 132, 10, wpack + WOFF_W1);
    pack_weights_kernel<<<50, 256, 0, stream>>>(W2, 132, 132, 10, wpack + WOFF_W2);
    pack_weights_kernel<<<40, 256, 0, stream>>>(W3, 132, 128,  8, wpack + WOFF_W3);
    pack_bias_kernel<<<1, 1024, 0, stream>>>(bm, bu, b0, b1, b2, b3, biasws);

    messageprop_fused_kernel<<<NROWS / 128, 256, SMEM_BYTES, stream>>>(
        signal, comps, wpack, biasws, (float*)d_out);
}
